// GRUDCell_76355928588490
// MI455X (gfx1250) — matrix operand order repaired and checked by reading, not compiled, not run on hardware
//
#include <hip/hip_runtime.h>
#include <math.h>

// ---------------------------------------------------------------------------
// GRU-D scan for MI455X (gfx1250, wave32, WMMA bf16 16x16x32, f32 accumulate)
//
// 16 WGs (one per 16-row batch tile) x 512 threads (16 waves). Each wave owns
// 16 hidden columns. Weights pre-packed to bf16 WMMA B-fragment layout
// (832 KB, L2-resident for all 512 steps). Per step:
//   phase0: gamma_dh GEMM (K=128)  -> h_d
//   phase1: [x|mf|h_d] @ [W1z|W1r] (K=512) -> z, r_pre
//   phase2: [x|mf|r*h_d] @ W2h     (K=512) -> hh -> h_t
// x/m tiles are double-buffered in LDS via async global->LDS copies
// (ASYNCcnt), issued one step ahead to hide HBM latency behind the GEMMs.
// Gate nonlinearities use native v_exp_f32 / v_rcp_f32 / v_tanh_f32.
// ---------------------------------------------------------------------------

typedef __attribute__((ext_vector_type(16))) __bf16 v16bf;
typedef __attribute__((ext_vector_type(8)))  float  v8f;

#define BB  256
#define TT  512
#define DD  128
#define HH  256
#define MT  16          // batch rows per workgroup
#define NTHREADS 512    // 16 waves

// packed bf16 weight buffers in workspace (elements)
#define W1_ELEMS (512*512)   // [Kz Kr; Mz Mr; Rz Rr]   K=512, N=512
#define W2_ELEMS (512*256)   // [Kh; Mh; Rh]            K=512, N=256
#define W0_ELEMS (128*256)   // hidden_decay_kernel     K=128, N=256
#define W1_OFF 0
#define W2_OFF (W1_ELEMS)
#define W0_OFF (W1_ELEMS + W2_ELEMS)

// dynamic LDS layout (byte offsets); no static __shared__ in the scan kernel
// so these are exact offsets from the wave's LDS base for async copies.
#define A1_OFF    0u                       // 16 KB  bf16[MT*512]
#define A2_OFF    16384u                   //  8 KB  bf16[MT*256]
#define A0_OFF    24576u                   //  4 KB  bf16[MT*128]
#define HPREV_OFF 28672u                   // 16 KB  f32 [MT*256]
#define XKEEP_OFF 45056u                   //  8 KB  f32 [MT*128]
#define SPREV_OFF 53248u                   //  8 KB  f32 [MT*128]
#define XBUF_OFF  61440u                   // 16 KB  f32 [2][MT*128]
#define MBUF_OFF  77824u                   // 16 KB  i32 [2][MT*128]
#define SMEM_BYTES 94208u                  // 92 KB total (< 320 KB/WGP)

// ---- fast transcendentals (single-instruction on gfx1250) -----------------
#if __has_builtin(__builtin_amdgcn_exp2f)
#define EXP2F __builtin_amdgcn_exp2f
#else
#define EXP2F exp2f
#endif
#if __has_builtin(__builtin_amdgcn_rcpf)
#define RCPF __builtin_amdgcn_rcpf
#else
#define RCPF(x) (1.f / (x))
#endif
__device__ __forceinline__ float fast_expe(float x) {   // e^x (args <= 0 here)
  return EXP2F(x * 1.44269504f);
}
__device__ __forceinline__ float fast_sigmoid(float x) {
  return RCPF(1.f + fast_expe(-x));
}
__device__ __forceinline__ float fast_tanh(float x) {
#if __has_builtin(__builtin_amdgcn_tanhf)
  return __builtin_amdgcn_tanhf(x);
#else
  float a = fabsf(x);
  float e = EXP2F(a * -2.88539008f);       // exp(-2a)
  float r = (1.f - e) * RCPF(1.f + e);
  return copysignf(r, x);
#endif
}

// ---- async global -> LDS copy (ASYNCcnt), 16B per lane --------------------
__device__ __forceinline__ void async_b128(unsigned lds_off, const void* gptr) {
  asm volatile("global_load_async_to_lds_b128 %0, %1, off"
               :: "v"(lds_off), "v"((unsigned long long)(uintptr_t)gptr)
               : "memory");
}
__device__ __forceinline__ void wait_async0() {
  asm volatile("s_wait_asynccnt 0x0" ::: "memory");
}

// --- WMMA A-matrix (16x32 bf16) fragment offset: element (m, kk) of tile kt.
__device__ __forceinline__ int a_off(int kt, int m, int kk) {
  int lane = m + ((kk >> 4) << 4);
  int idx  = kk & 15;
  return ((kt << 5) + lane) * 16 + idx;
}

__device__ __forceinline__ v8f wmma_bf16(v16bf a, v16bf b, v8f c) {
  return __builtin_amdgcn_wmma_f32_16x16x32_bf16(
      /*neg_a=*/false, a, /*neg_b=*/false, b,
      /*c_mod=*/(short)0, c, /*reuse_a=*/false, /*reuse_b=*/false);
}

// ---------------------------------------------------------------------------
// One-shot weight pack: f32 row-major -> bf16 WMMA B-fragment tiles.
// B-tile (32K x 16N): lane = n%16 + 16*(k%32 >= 16); idx = k & 15.
// ---------------------------------------------------------------------------
__global__ void grud_pack_weights(const float* __restrict__ kern,
                                  const float* __restrict__ rk,
                                  const float* __restrict__ mk,
                                  const float* __restrict__ hdk,
                                  __bf16* __restrict__ ws) {
  const int total = W1_ELEMS + W2_ELEMS + W0_ELEMS;
  for (int p = blockIdx.x * blockDim.x + threadIdx.x; p < total;
       p += gridDim.x * blockDim.x) {
    float v;
    if (p < W1_ELEMS) {                       // z|r columns (0..511)
      int q = p;
      int idx = q & 15, lane = (q >> 4) & 31, rest = q >> 9;
      int nt = rest & 31, kt = rest >> 5;
      int k = (kt << 5) + ((lane >> 4) << 4) + idx;
      int n = (nt << 4) + (lane & 15);
      v = (k < 128) ? kern[k * 768 + n]
        : (k < 256) ? mk[(k - 128) * 768 + n]
                    : rk[(k - 256) * 768 + n];
      ws[W1_OFF + q] = (__bf16)v;
    } else if (p < W1_ELEMS + W2_ELEMS) {     // h columns (512..767)
      int q = p - W1_ELEMS;
      int idx = q & 15, lane = (q >> 4) & 31, rest = q >> 9;
      int nt = rest & 15, kt = rest >> 4;
      int k = (kt << 5) + ((lane >> 4) << 4) + idx;
      int n = 512 + (nt << 4) + (lane & 15);
      v = (k < 128) ? kern[k * 768 + n]
        : (k < 256) ? mk[(k - 128) * 768 + n]
                    : rk[(k - 256) * 768 + n];
      ws[W2_OFF + q] = (__bf16)v;
    } else {                                  // hidden decay kernel
      int q = p - W1_ELEMS - W2_ELEMS;
      int idx = q & 15, lane = (q >> 4) & 31, rest = q >> 9;
      int nt = rest & 15, kt = rest >> 4;
      int k = (kt << 5) + ((lane >> 4) << 4) + idx;
      int n = (nt << 4) + (lane & 15);
      v = hdk[k * 256 + n];
      ws[W0_OFF + q] = (__bf16)v;
    }
  }
}

// ---------------------------------------------------------------------------
// Persistent recurrent kernel.
// ---------------------------------------------------------------------------
__global__ __launch_bounds__(NTHREADS)
void grud_scan(const float* __restrict__ x, const int* __restrict__ msk,
               const float* __restrict__ s, const float* __restrict__ bias,
               const float* __restrict__ idk, const float* __restrict__ idb,
               const float* __restrict__ hdb, const __bf16* __restrict__ wsb,
               float* __restrict__ out) {
  extern __shared__ __align__(32) char smem[];
  __bf16* A1    = (__bf16*)(smem + A1_OFF);    // K: [0..127]=x [128..255]=mf [256..511]=h_d
  __bf16* A2    = (__bf16*)(smem + A2_OFF);    // r_pre * h_d
  __bf16* A0    = (__bf16*)(smem + A0_OFF);    // d_t
  float*  hprev = (float*)(smem + HPREV_OFF);
  float*  xkeep = (float*)(smem + XKEEP_OFF);
  float*  sprev = (float*)(smem + SPREV_OFF);
  float*  xbuf  = (float*)(smem + XBUF_OFF);   // [2][MT*DD]
  int*    mbuf  = (int*)(smem + MBUF_OFF);     // [2][MT*DD]

  const int tid   = threadIdx.x;
  const int lane  = tid & 31;
  const int w     = tid >> 5;        // wave id 0..15
  const int nloc  = lane & 15;       // N within C/D tile
  const int hi    = lane >> 4;       // M = v + 8*hi
  const int bbase = blockIdx.x * MT;

  // per-thread async chunk: row pm, 4 floats starting at column pc
  const int pm = tid >> 5, pc = (tid & 31) << 2;
  const unsigned pl = (unsigned)((pm * DD + pc) * 4);

  // prime t=0 tile
  {
    size_t gi = (((size_t)(bbase + pm)) * TT + 0) * DD + pc;
    async_b128(XBUF_OFF + pl, x + gi);
    async_b128(MBUF_OFF + pl, msk + gi);
  }

  for (int i = tid; i < MT * HH; i += NTHREADS) hprev[i] = 0.f;
  for (int i = tid; i < MT * DD; i += NTHREADS) { xkeep[i] = 0.f; sprev[i] = 0.f; }

  const int   j    = (w << 4) + nloc;          // owned hidden column
  const float bz   = bias[j];
  const float br   = bias[HH + j];
  const float bh   = bias[2 * HH + j];
  const float hdbj = hdb[j];

  for (int t = 0; t < TT; ++t) {
    const int buf = t & 1;
    wait_async0();                 // this wave's async copies for step t done
    __syncthreads();               // everyone's copies visible

    // ---------------- stage A: elementwise inputs + A-fragment staging -----
    #pragma unroll
    for (int ii = 0; ii < 4; ++ii) {
      int e  = ii * NTHREADS + tid;            // 0..2047
      int mm = e >> 7, d = e & 127;
      int b  = bbase + mm;
      float xt = xbuf[buf * (MT * DD) + mm * DD + d];
      int   mi = mbuf[buf * (MT * DD) + mm * DD + d];
      float st = s[(size_t)b * TT + t];
      float sp = sprev[mm * DD + d];
      float dt = st - sp;
      float gdi = fast_expe(-fmaxf(dt * idk[d] + idb[d], 0.f));
      float xk  = xkeep[mm * DD + d];
      float xkn = mi ? xt : xk;
      float xv  = mi ? xt : gdi * xkn;
      xkeep[mm * DD + d] = xkn;
      sprev[mm * DD + d] = mi ? st : sp;
      A1[a_off(d >> 5, mm, d & 31)] = (__bf16)xv;
      int kc = 128 + d;
      A1[a_off(kc >> 5, mm, kc & 31)] = (__bf16)(float)mi;
      A0[a_off(d >> 5, mm, d & 31)] = (__bf16)dt;
    }

    // prefetch step t+1 while the GEMM phases run
    if (t + 1 < TT) {
      size_t gi = (((size_t)(bbase + pm)) * TT + (t + 1)) * DD + pc;
      unsigned boff = (unsigned)((t + 1) & 1) * (MT * DD * 4);
      async_b128(XBUF_OFF + boff + pl, x + gi);
      async_b128(MBUF_OFF + boff + pl, msk + gi);
    }
    __syncthreads();

    // ---------------- phase 0: gamma_dh = exp(-relu(d_t @ Wdh + bdh)) ------
    v8f acc0 = {};
    #pragma unroll
    for (int kt = 0; kt < 4; ++kt) {
      v16bf a = *(const v16bf*)(A0 + (((kt << 5) + lane) << 4));
      v16bf b = *(const v16bf*)(wsb + W0_OFF + ((((kt << 4) + w) * 32 + lane) << 4));
      acc0 = wmma_bf16(a, b, acc0);
    }
    #pragma unroll
    for (int v = 0; v < 8; ++v) {
      int   mm = v + (hi << 3);
      float g  = fast_expe(-fmaxf(acc0[v] + hdbj, 0.f));
      float hd = g * hprev[mm * HH + j];
      int   kc = 256 + j;
      A1[a_off(kc >> 5, mm, kc & 31)] = (__bf16)hd;   // h_d into fused A1
    }
    __syncthreads();

    // ---------------- phase 1: [x|mf|h_d] @ [W1z | W1r]  (K=512) -----------
    v8f accZ = {}, accR = {};
    #pragma unroll
    for (int kt = 0; kt < 16; ++kt) {
      v16bf a   = *(const v16bf*)(A1 + (((kt << 5) + lane) << 4));
      v16bf bzf = *(const v16bf*)(wsb + W1_OFF + ((((kt << 5) + w) * 32 + lane) << 4));
      v16bf brf = *(const v16bf*)(wsb + W1_OFF + ((((kt << 5) + 16 + w) * 32 + lane) << 4));
      accZ = wmma_bf16(a, bzf, accZ);
      accR = wmma_bf16(a, brf, accR);
    }
    float zreg[8];
    #pragma unroll
    for (int v = 0; v < 8; ++v) {
      int   mm = v + (hi << 3);
      zreg[v]  = fast_sigmoid(accZ[v] + bz);
      float rp = accR[v] + br;                         // reference: raw r_pre
      int   kc = 256 + j;
      float hd = (float)A1[a_off(kc >> 5, mm, kc & 31)];
      A2[a_off(j >> 5, mm, j & 31)] = (__bf16)(rp * hd);
    }
    __syncthreads();

    // ---------------- phase 2: [x|mf|r*h_d] @ W2h  (K=512) -----------------
    v8f accH = {};
    #pragma unroll
    for (int kt = 0; kt < 16; ++kt) {
      const __bf16* asrc = (kt < 8)
          ? (A1 + (((kt << 5) + lane) << 4))
          : (A2 + ((((kt - 8) << 5) + lane) << 4));
      v16bf a = *(const v16bf*)asrc;
      v16bf b = *(const v16bf*)(wsb + W2_OFF + ((((kt << 4) + w) * 32 + lane) << 4));
      accH = wmma_bf16(a, b, accH);
    }
    #pragma unroll
    for (int v = 0; v < 8; ++v) {
      int   mm = v + (hi << 3);
      float hh = fast_tanh(accH[v] + bh);
      float hp = hprev[mm * HH + j];
      float ht = zreg[v] * hp + (1.f - zreg[v]) * hh;
      hprev[mm * HH + j] = ht;
      out[(((size_t)(bbase + mm)) * TT + t) * HH + j] = ht;
    }
    __syncthreads();
  }
}

extern "C" void kernel_launch(void* const* d_in, const int* in_sizes, int n_in,
                              void* d_out, int out_size, void* d_ws, size_t ws_size,
                              hipStream_t stream) {
  (void)in_sizes; (void)n_in; (void)out_size; (void)ws_size;
  const float* x    = (const float*)d_in[0];
  const int*   m    = (const int*)d_in[1];
  const float* s    = (const float*)d_in[2];
  const float* kern = (const float*)d_in[3];
  const float* rk   = (const float*)d_in[4];
  const float* mk   = (const float*)d_in[5];
  const float* bias = (const float*)d_in[6];
  const float* idk  = (const float*)d_in[7];
  const float* idb  = (const float*)d_in[8];
  const float* hdk  = (const float*)d_in[9];
  const float* hdb  = (const float*)d_in[10];
  __bf16* ws = (__bf16*)d_ws;

  // pack f32 weights -> bf16 WMMA-fragment layout (832 KB of workspace)
  grud_pack_weights<<<1664, 256, 0, stream>>>(kern, rk, mk, hdk, ws);
  // persistent recurrent scan: 16 WGs x 512 threads, 92 KB dynamic LDS
  grud_scan<<<BB / MT, NTHREADS, SMEM_BYTES, stream>>>(
      x, m, s, bias, idk, idb, hdb, ws, (float*)d_out);
}
